// GATModel_91182155694572
// MI455X (gfx1250) — matrix operand order repaired and checked by reading, not compiled, not run on hardware
//
#include <hip/hip_runtime.h>
#include <hip/hip_bf16.h>

// ---------------- problem constants (match reference) ----------------
#define N_NODES   40000
#define E_EDGES   400000
#define ET_EDGES  (E_EDGES + N_NODES)   // edges + self loops = 440000
#define F_IN      128
#define H_HEADS   4
#define C_CH      64
#define HC        256
#define G_GRAPHS  128
#define NC_CLASSES 10
#define MT_TILES  (N_NODES / 16)        // 2500
#define NT_TILES  (HC / 16)             // 16
#define NEG_SLOPE 0.2f

typedef __attribute__((ext_vector_type(16))) __bf16 v16bf;
typedef __attribute__((ext_vector_type(8)))  float  v8f;

// ---- order-preserving float<->uint key for atomicMax-based segment max ----
__device__ __forceinline__ unsigned f2key(float f) {
    unsigned u = __float_as_uint(f);
    return (u & 0x80000000u) ? ~u : (u | 0x80000000u);
}
__device__ __forceinline__ float key2f(unsigned k) {
    unsigned u = (k & 0x80000000u) ? (k ^ 0x80000000u) : ~k;
    return __uint_as_float(u);
}
__device__ __forceinline__ float lrelu(float v) {
    return v > 0.0f ? v : NEG_SLOPE * v;
}

// =====================================================================
// Pack activation [N,K] f32 (row-major) into WMMA A-fragment bf16 layout
// A-frag (16-bit, 16x32): lane 0-15 -> M=lane, lanes 16-31 -> M=lane-16.
// elem j (j<8):  k = klo + j        (klo = (lane>=16) ? 8 : 0)
// elem j (j>=8): k = 16 + klo + j-8
// Stored as [mt][kt][lane][16] contiguous (32B per lane chunk).
// =====================================================================
__global__ void pack_a_kernel(const float* __restrict__ X, __bf16* __restrict__ A,
                              int K, int KT) {
    int tid = blockIdx.x * blockDim.x + threadIdx.x;
    int total = MT_TILES * KT * 32;
    if (tid >= total) return;
    int lane = tid & 31;
    int kt   = (tid >> 5) % KT;
    int mt   = tid / (KT * 32);
    int m    = mt * 16 + (lane & 15);
    int klo  = (lane & 16) ? 8 : 0;
    const float* row = X + (size_t)m * K + kt * 32;
    float4 f0 = *(const float4*)(row + klo);
    float4 f1 = *(const float4*)(row + klo + 4);
    float4 f2 = *(const float4*)(row + 16 + klo);
    float4 f3 = *(const float4*)(row + 16 + klo + 4);
    v16bf v;
    v[0]=(__bf16)f0.x;  v[1]=(__bf16)f0.y;  v[2]=(__bf16)f0.z;  v[3]=(__bf16)f0.w;
    v[4]=(__bf16)f1.x;  v[5]=(__bf16)f1.y;  v[6]=(__bf16)f1.z;  v[7]=(__bf16)f1.w;
    v[8]=(__bf16)f2.x;  v[9]=(__bf16)f2.y;  v[10]=(__bf16)f2.z; v[11]=(__bf16)f2.w;
    v[12]=(__bf16)f3.x; v[13]=(__bf16)f3.y; v[14]=(__bf16)f3.z; v[15]=(__bf16)f3.w;
    *(v16bf*)(A + ((size_t)(mt * KT + kt) * 32 + lane) * 16) = v;
}

// =====================================================================
// Pack weight [K,HC] f32 into WMMA B-fragment bf16 layout (32x16 tile):
// lanes 0-15: col = lane, K = kt*32 + 0..7,16..23 ; lanes 16-31: K = kt*32 + 8..15,24..31
// elem j -> k offset j (j<8) or j+8 (j>=8), plus 8 for lanes 16-31 (2 K values per VGPR dword).
// =====================================================================
__global__ void pack_b_kernel(const float* __restrict__ W, __bf16* __restrict__ B,
                              int K, int KT) {
    int tid = blockIdx.x * blockDim.x + threadIdx.x;
    int total = NT_TILES * KT * 32;
    if (tid >= total) return;
    int lane = tid & 31;
    int kt   = (tid >> 5) % KT;
    int nt   = tid / (KT * 32);
    int n    = nt * 16 + (lane & 15);
    int kb   = kt * 32 + ((lane & 16) ? 8 : 0);
    v16bf v;
#pragma unroll
    for (int j = 0; j < 16; ++j)
        v[j] = (__bf16)W[(size_t)(kb + j + (j & 8)) * HC + n];
    *(v16bf*)(B + ((size_t)(nt * KT + kt) * 32 + lane) * 16) = v;
}

// =====================================================================
// GEMM: C[N,HC] = A[N,K] * W[K,HC], bf16 inputs, f32 accumulate, WMMA.
// One block = 8 waves; wave w computes output tiles (mt, 2w) and (mt, 2w+1).
// =====================================================================
__global__ void __launch_bounds__(256)
gemm_wmma_kernel(const __bf16* __restrict__ A, const __bf16* __restrict__ B,
                 float* __restrict__ C, int KT) {
    int mt   = blockIdx.x;
    int wave = threadIdx.x >> 5;
    int lane = threadIdx.x & 31;
    int nt0  = wave * 2;
    int nt1  = nt0 + 1;
    const __bf16* ap  = A + ((size_t)mt  * KT * 32 + lane) * 16;
    const __bf16* bp0 = B + ((size_t)nt0 * KT * 32 + lane) * 16;
    const __bf16* bp1 = B + ((size_t)nt1 * KT * 32 + lane) * 16;
    v8f acc0 = {}; v8f acc1 = {};
    for (int kt = 0; kt < KT; ++kt) {
        __builtin_prefetch(ap + 512, 0, 0);       // global_prefetch_b8 next K-tile
        v16bf a  = *(const v16bf*)ap;  ap  += 512;
        v16bf b0 = *(const v16bf*)bp0; bp0 += 512;
        v16bf b1 = *(const v16bf*)bp1; bp1 += 512;
        acc0 = __builtin_amdgcn_wmma_f32_16x16x32_bf16(false, a, false, b0,
                                                       (short)0, acc0, false, false);
        acc1 = __builtin_amdgcn_wmma_f32_16x16x32_bf16(false, a, false, b1,
                                                       (short)0, acc1, false, false);
    }
    // C/D layout: VGPR r -> row r (lanes 0-15) / row r+8 (lanes 16-31), col=lane%16
    int col  = lane & 15;
    int mrow = mt * 16 + ((lane & 16) ? 8 : 0);
#pragma unroll
    for (int r = 0; r < 8; ++r) {
        C[(size_t)(mrow + r) * HC + nt0 * 16 + col] = acc0[r];
        C[(size_t)(mrow + r) * HC + nt1 * 16 + col] = acc1[r];
    }
}

// ===================== per-node attention logits =====================
__global__ void logits_kernel(const float* __restrict__ Hb,
                              const float* __restrict__ Asrc,
                              const float* __restrict__ Adst,
                              float* __restrict__ ls, float* __restrict__ ld) {
    int tid = blockIdx.x * blockDim.x + threadIdx.x;
    if (tid >= N_NODES * H_HEADS) return;
    int hh = tid & 3;
    int n  = tid >> 2;
    const float4* hp = (const float4*)(Hb + (size_t)n * HC + hh * C_CH);
    const float4* sp = (const float4*)(Asrc + hh * C_CH);
    const float4* dp = (const float4*)(Adst + hh * C_CH);
    float ss = 0.0f, sd = 0.0f;
#pragma unroll 4
    for (int i = 0; i < C_CH / 4; ++i) {
        float4 h = hp[i], a = sp[i], d = dp[i];
        ss += h.x * a.x + h.y * a.y + h.z * a.z + h.w * a.w;
        sd += h.x * d.x + h.y * d.y + h.z * d.z + h.w * d.w;
    }
    ls[tid] = ss;
    ld[tid] = sd;
}

// ============ init: out = bias (broadcast), maxkey = 0, denom = 1e-16 ============
__global__ void init_layer_kernel(float* __restrict__ Out, const float* __restrict__ bias,
                                  unsigned* __restrict__ maxk, float* __restrict__ den) {
    int tid = blockIdx.x * blockDim.x + threadIdx.x;
    int total = N_NODES * HC / 4;
    if (tid < total) {
        int col4 = tid & (HC / 4 - 1);
        ((float4*)Out)[tid] = ((const float4*)bias)[col4];
    }
    if (tid < N_NODES * H_HEADS) {
        maxk[tid] = 0u;          // encodes "below any finite float"
        den[tid]  = 1e-16f;      // reference adds 1e-16 to segment_sum
    }
}

__device__ __forceinline__ void edge_sd(int e, const int* __restrict__ src,
                                        const int* __restrict__ dst, int& s, int& d) {
    if (e < E_EDGES) { s = src[e]; d = dst[e]; }
    else             { s = d = e - E_EDGES; }   // self loop
}

// ===================== segment max over dst neighborhoods =====================
__global__ void edge_max_kernel(const int* __restrict__ src, const int* __restrict__ dst,
                                const float* __restrict__ ls, const float* __restrict__ ld,
                                unsigned* __restrict__ maxk) {
    int e = blockIdx.x * blockDim.x + threadIdx.x;
    if (e >= ET_EDGES) return;
    int s, d; edge_sd(e, src, dst, s, d);
#pragma unroll
    for (int hh = 0; hh < H_HEADS; ++hh) {
        float v = lrelu(ls[s * H_HEADS + hh] + ld[d * H_HEADS + hh]);
        atomicMax(&maxk[d * H_HEADS + hh], f2key(v));
    }
}

// ===================== softmax denominator =====================
__global__ void edge_denom_kernel(const int* __restrict__ src, const int* __restrict__ dst,
                                  const float* __restrict__ ls, const float* __restrict__ ld,
                                  const unsigned* __restrict__ maxk, float* __restrict__ den) {
    int e = blockIdx.x * blockDim.x + threadIdx.x;
    if (e >= ET_EDGES) return;
    int s, d; edge_sd(e, src, dst, s, d);
#pragma unroll
    for (int hh = 0; hh < H_HEADS; ++hh) {
        float v = lrelu(ls[s * H_HEADS + hh] + ld[d * H_HEADS + hh]);
        float w = __expf(v - key2f(maxk[d * H_HEADS + hh]));
        atomicAdd(&den[d * H_HEADS + hh], w);
    }
}

// ===================== normalized attention coefficients =====================
__global__ void edge_alpha_kernel(const int* __restrict__ src, const int* __restrict__ dst,
                                  const float* __restrict__ ls, const float* __restrict__ ld,
                                  const unsigned* __restrict__ maxk,
                                  const float* __restrict__ den, float* __restrict__ alpha) {
    int tid = blockIdx.x * blockDim.x + threadIdx.x;
    if (tid >= ET_EDGES * H_HEADS) return;
    int hh = tid & 3;
    int e  = tid >> 2;
    int s, d; edge_sd(e, src, dst, s, d);
    float v = lrelu(ls[s * H_HEADS + hh] + ld[d * H_HEADS + hh]);
    float w = __expf(v - key2f(maxk[d * H_HEADS + hh]));
    alpha[tid] = w / den[d * H_HEADS + hh];
}

// ============ weighted scatter-add: out[dst] += h[src] * alpha ============
// 64 threads per edge, float4 each (HC=256). L2-resident atomics.
__global__ void scatter_kernel(const int* __restrict__ src, const int* __restrict__ dst,
                               const float* __restrict__ Hb, const float* __restrict__ alpha,
                               float* __restrict__ Out) {
    int tid = blockIdx.x * blockDim.x + threadIdx.x;
    if (tid >= ET_EDGES * 64) return;
    int t = tid & 63;
    int e = tid >> 6;
    int s, d; edge_sd(e, src, dst, s, d);
    float a = alpha[e * H_HEADS + (t >> 4)];     // head = channel4 / 16
    float4 hv = *(const float4*)(Hb + (size_t)s * HC + t * 4);
    float* op = Out + (size_t)d * HC + t * 4;
    atomicAdd(op + 0, hv.x * a);
    atomicAdd(op + 1, hv.y * a);
    atomicAdd(op + 2, hv.z * a);
    atomicAdd(op + 3, hv.w * a);
}

// ===================== global add pool =====================
__global__ void pool_init_kernel(float* __restrict__ pooled) {
    int tid = blockIdx.x * blockDim.x + threadIdx.x;
    if (tid < G_GRAPHS * HC) pooled[tid] = 0.0f;
}
__global__ void pool_kernel(const float* __restrict__ Xf, const int* __restrict__ batch,
                            float* __restrict__ pooled) {
    int tid = blockIdx.x * blockDim.x + threadIdx.x;
    if (tid >= N_NODES * 64) return;
    int t = tid & 63;
    int n = tid >> 6;
    int g = batch[n];
    float4 v = *(const float4*)(Xf + (size_t)n * HC + t * 4);
    float* op = pooled + (size_t)g * HC + t * 4;
    atomicAdd(op + 0, v.x);
    atomicAdd(op + 1, v.y);
    atomicAdd(op + 2, v.z);
    atomicAdd(op + 3, v.w);
}

// ===================== final FC =====================
__global__ void fc_kernel(const float* __restrict__ pooled, const float* __restrict__ fcW,
                          const float* __restrict__ fcb, float* __restrict__ out) {
    int tid = blockIdx.x * blockDim.x + threadIdx.x;
    if (tid >= G_GRAPHS * NC_CLASSES) return;
    int g = tid / NC_CLASSES;
    int j = tid % NC_CLASSES;
    float s = fcb[j];
    for (int k = 0; k < HC; ++k)
        s += pooled[(size_t)g * HC + k] * fcW[(size_t)k * NC_CLASSES + j];
    out[tid] = s;
}

// =====================================================================
static inline size_t alg(size_t x) { return (x + 255) & ~(size_t)255; }

extern "C" void kernel_launch(void* const* d_in, const int* in_sizes, int n_in,
                              void* d_out, int out_size, void* d_ws, size_t ws_size,
                              hipStream_t stream) {
    (void)in_sizes; (void)n_in; (void)out_size; (void)ws_size;
    const float* x     = (const float*)d_in[0];
    const int*   ei    = (const int*)d_in[1];
    const int*   batch = (const int*)d_in[2];
    const float* W[3]  = {(const float*)d_in[3], (const float*)d_in[7],  (const float*)d_in[11]};
    const float* As[3] = {(const float*)d_in[4], (const float*)d_in[8],  (const float*)d_in[12]};
    const float* Ad[3] = {(const float*)d_in[5], (const float*)d_in[9],  (const float*)d_in[13]};
    const float* Bi[3] = {(const float*)d_in[6], (const float*)d_in[10], (const float*)d_in[14]};
    const float* fcW   = (const float*)d_in[15];
    const float* fcb   = (const float*)d_in[16];
    const int* srcp = ei;
    const int* dstp = ei + E_EDGES;

    // ---- workspace partition ----
    char* ws = (char*)d_ws;
    size_t o = 0;
    __bf16* Abuf = (__bf16*)(ws + o); o += alg((size_t)N_NODES * HC * 2);   // bf16 A frags (max K=256)
    __bf16* Wb   = (__bf16*)(ws + o); o += alg((size_t)HC * HC * 2);        // bf16 B frags
    float* bufA  = (float*)(ws + o);  o += alg((size_t)N_NODES * HC * 4);
    float* bufB  = (float*)(ws + o);  o += alg((size_t)N_NODES * HC * 4);
    float* ls    = (float*)(ws + o);  o += alg((size_t)N_NODES * H_HEADS * 4);
    float* ld    = (float*)(ws + o);  o += alg((size_t)N_NODES * H_HEADS * 4);
    unsigned* mk = (unsigned*)(ws + o); o += alg((size_t)N_NODES * H_HEADS * 4);
    float* den   = (float*)(ws + o);  o += alg((size_t)N_NODES * H_HEADS * 4);
    float* alpha = (float*)(ws + o);  o += alg((size_t)ET_EDGES * H_HEADS * 4);
    float* pooled= (float*)(ws + o);  o += alg((size_t)G_GRAPHS * HC * 4);

    const float* cur = x;
    float* hbuf = bufA;
    float* obuf = bufB;
    const int BT = 256;

    for (int L = 0; L < 3; ++L) {
        int K  = (L == 0) ? F_IN : HC;
        int KT = K / 32;
        pack_b_kernel<<<(NT_TILES * KT * 32 + BT - 1) / BT, BT, 0, stream>>>(W[L], Wb, K, KT);
        pack_a_kernel<<<(MT_TILES * KT * 32 + BT - 1) / BT, BT, 0, stream>>>(cur, Abuf, K, KT);
        gemm_wmma_kernel<<<MT_TILES, BT, 0, stream>>>(Abuf, Wb, hbuf, KT);
        logits_kernel<<<(N_NODES * H_HEADS + BT - 1) / BT, BT, 0, stream>>>(hbuf, As[L], Ad[L], ls, ld);
        init_layer_kernel<<<(N_NODES * HC / 4 + BT - 1) / BT, BT, 0, stream>>>(obuf, Bi[L], mk, den);
        edge_max_kernel<<<(ET_EDGES + BT - 1) / BT, BT, 0, stream>>>(srcp, dstp, ls, ld, mk);
        edge_denom_kernel<<<(ET_EDGES + BT - 1) / BT, BT, 0, stream>>>(srcp, dstp, ls, ld, mk, den);
        edge_alpha_kernel<<<(ET_EDGES * H_HEADS + BT - 1) / BT, BT, 0, stream>>>(srcp, dstp, ls, ld, mk, den, alpha);
        scatter_kernel<<<(ET_EDGES * 64 + BT - 1) / BT, BT, 0, stream>>>(srcp, dstp, hbuf, alpha, obuf);
        cur = obuf;
        float* t = hbuf; hbuf = obuf; obuf = t;   // ping-pong
    }

    pool_init_kernel<<<(G_GRAPHS * HC + BT - 1) / BT, BT, 0, stream>>>(pooled);
    pool_kernel<<<(N_NODES * 64 + BT - 1) / BT, BT, 0, stream>>>(cur, batch, pooled);
    fc_kernel<<<(G_GRAPHS * NC_CLASSES + BT - 1) / BT, BT, 0, stream>>>(pooled, fcW, fcb, (float*)d_out);
}
